// UnifiedMambaBlock_69166153335221
// MI455X (gfx1250) — matrix operand order repaired and checked by reading, not compiled, not run on hardware
//
#include <hip/hip_runtime.h>
#include <hip/hip_bf16.h>

// ---------------------------------------------------------------------------
// Mamba block for MI455X (gfx1250, wave32, WMMA).
// Heavy GEMMs: bf16 v_wmma_f32_16x16x32_bf16, f32 accumulate.
// 128x128 block tile, BK=64, 8 waves * (32x64) => 8 accumulators/wave.
// Tiles staged with GLOBAL_LOAD_ASYNC_TO_LDS_B128 (ASYNCcnt) + double buffer.
// ---------------------------------------------------------------------------

typedef __attribute__((ext_vector_type(16))) __bf16 v16bf;
typedef __attribute__((ext_vector_type(8)))  float  v8f;
typedef __attribute__((ext_vector_type(4)))  int    async_v4i;

#define D_MODEL  2048
#define D_STATE  16
#define D_CONV   4
#define D_INNER  4096
#define DT_RANK  128
#define B_SZ     2
#define SEQ      1024
#define M_ROWS   (B_SZ * SEQ)          // 2048
#define BC_STRIDE 128                  // padded B/C projection width

#if __has_builtin(__builtin_amdgcn_global_load_async_to_lds_b128)
#define HAVE_ASYNC_LDS 1
// param 1: global (addrspace(1)) int4* source; param 2: LDS (addrspace(3)) dst
#define ASYNC_B128(g, l)                                                  \
    __builtin_amdgcn_global_load_async_to_lds_b128(                       \
        (__attribute__((address_space(1))) async_v4i*)(g),                \
        (__attribute__((address_space(3))) async_v4i*)(l), 0, 0)
#else
#define HAVE_ASYNC_LDS 0
#endif

#if __has_builtin(__builtin_amdgcn_s_wait_asynccnt)
#define WAIT_ASYNC(n) __builtin_amdgcn_s_wait_asynccnt(n)
#else
#define WAIT_ASYNC(n) asm volatile("s_wait_asynccnt %0" :: "i"(n) : "memory")
#endif

__device__ __forceinline__ unsigned short f2bf(float f) {
    union { float f; unsigned u; } v; v.f = f;
    unsigned r = v.u + 0x7fffu + ((v.u >> 16) & 1u);   // round-to-nearest-even
    return (unsigned short)(r >> 16);
}

__device__ __forceinline__ float silu_f(float x) {
    return x / (1.0f + __expf(-x));
}

// ---------------------------------------------------------------------------
// f32 -> bf16 conversion
// ---------------------------------------------------------------------------
__global__ void cvt_bf16_kernel(const float* __restrict__ src,
                                unsigned short* __restrict__ dst, int n) {
    int i = blockIdx.x * 256 + threadIdx.x;
    if (i < n) dst[i] = f2bf(src[i]);
}

// Packed [128, 4096] weight: rows 0..15 = w_xb, 16..31 = w_xc, 32..127 = 0
__global__ void build_wbc_kernel(const float* __restrict__ wxb,
                                 const float* __restrict__ wxc,
                                 unsigned short* __restrict__ dst) {
    int i = blockIdx.x * 256 + threadIdx.x;       // 128*4096 elements
    int row = i >> 12;
    int col = i & 4095;
    float v = 0.0f;
    if (row < 16)       v = wxb[row * D_INNER + col];
    else if (row < 32)  v = wxc[(row - 16) * D_INNER + col];
    dst[i] = f2bf(v);
}

// ---------------------------------------------------------------------------
// WMMA GEMM:  C[M,N] = A[M,K] @ W[N,K]^T (+ bias[N])
// Block tile 128x128, BK=64, 256 threads = 8 waves (4 M-waves x 2 N-waves).
// Wave tile 32x64 -> 8 f32 accumulators, 16 wmma per K-iteration.
// ---------------------------------------------------------------------------
#define GEMM_BK   64
#define LDS_STR   72     // 64 + 8 bf16 pad (16B) -> conflict-free b128 reads

__global__ __launch_bounds__(256) void wmma_gemm_bf16(
    const unsigned short* __restrict__ A,   // [M,K] bf16 row-major
    const unsigned short* __restrict__ W,   // [N,K] bf16 row-major
    const float* __restrict__ bias,         // [N] or nullptr
    float* __restrict__ C,                  // [M,N] f32 row-major
    int M, int N, int K)
{
#if HAVE_ASYNC_LDS
    __shared__ unsigned short sA[2][128 * LDS_STR];
    __shared__ unsigned short sW[2][128 * LDS_STR];
#else
    __shared__ unsigned short sA[1][128 * LDS_STR];
    __shared__ unsigned short sW[1][128 * LDS_STR];
#endif

    const int t    = threadIdx.x;
    const int lane = t & 31;
    const int wv   = t >> 5;
    const int mw   = wv >> 1;       // 0..3  (rows mw*32 .. +32)
    const int nw   = wv & 1;        // 0..1  (cols nw*64 .. +64)
    const int m0   = blockIdx.y * 128;
    const int n0   = blockIdx.x * 128;

    // cooperative loader: 128 rows x 64 cols per tile, 32 bf16 (4xb128)/thread
    const int lrow = t >> 1;            // 0..127
    const int lseg = (t & 1) << 5;      // 0 or 32 (elements)

    // fragment lane mapping (ISA 7.12.2, bf16 16x32 A / 32x16 B, wave32)
    const int frow = lane & 15;
    const int kh   = lane >> 4;

    v8f acc[2][4];
    #pragma unroll
    for (int ms = 0; ms < 2; ++ms)
        #pragma unroll
        for (int ns = 0; ns < 4; ++ns)
            acc[ms][ns] = (v8f){};

    union Frag { v16bf v; uint4 q[2]; };

    // compute one BK=64 tile from LDS buffer `buf`
    auto compute_tile = [&](int buf) {
        #pragma unroll
        for (int ks = 0; ks < 2; ++ks) {           // two 32-wide K substeps
            const int kb = ks * 32;
            Frag fa[2], fb[4];
            // A: lanes 0-15 K {0..7,16..23}, lanes 16-31 K {8..15,24..31}
            #pragma unroll
            for (int ms = 0; ms < 2; ++ms) {
                const unsigned short* pa =
                    &sA[buf][(mw * 32 + ms * 16 + frow) * LDS_STR + kb];
                fa[ms].q[0] = *(const uint4*)&pa[kh * 8];
                fa[ms].q[1] = *(const uint4*)&pa[16 + kh * 8];
            }
            // B: lane n = lane&15; lanes 0-15 K {0..7,16..23}, lanes 16-31 K {8..15,24..31}
            #pragma unroll
            for (int ns = 0; ns < 4; ++ns) {
                const unsigned short* pb =
                    &sW[buf][(nw * 64 + ns * 16 + frow) * LDS_STR + kb];
                fb[ns].q[0] = *(const uint4*)&pb[kh * 8];
                fb[ns].q[1] = *(const uint4*)&pb[16 + kh * 8];
            }
            #pragma unroll
            for (int ms = 0; ms < 2; ++ms)
                #pragma unroll
                for (int ns = 0; ns < 4; ++ns)
                    acc[ms][ns] = __builtin_amdgcn_wmma_f32_16x16x32_bf16(
                        false, fa[ms].v, false, fb[ns].v,
                        (short)0, acc[ms][ns], false, false);
        }
    };

#if HAVE_ASYNC_LDS
    // async Global->LDS, double-buffered: 8 b128 async ops per thread per tile
    auto issue_tile = [&](int k0, int buf) {
        const unsigned short* gA = &A[(size_t)(m0 + lrow) * K + k0 + lseg];
        const unsigned short* gW = &W[(size_t)(n0 + lrow) * K + k0 + lseg];
        unsigned short* lA = &sA[buf][lrow * LDS_STR + lseg];
        unsigned short* lW = &sW[buf][lrow * LDS_STR + lseg];
        #pragma unroll
        for (int s = 0; s < 4; ++s) {
            ASYNC_B128(gA + s * 8, lA + s * 8);
            ASYNC_B128(gW + s * 8, lW + s * 8);
        }
    };

    const int nk = K / GEMM_BK;
    issue_tile(0, 0);
    for (int kt = 0; kt < nk; ++kt) {
        const int cur = kt & 1;
        if (kt + 1 < nk) {
            issue_tile((kt + 1) * GEMM_BK, cur ^ 1);  // overlap with compute
            WAIT_ASYNC(8);   // only newest tile's 8 ops may stay in flight
        } else {
            WAIT_ASYNC(0);
        }
        __syncthreads();     // tile kt visible to all waves
        compute_tile(cur);
        __syncthreads();     // all waves done reading buf before reuse
    }
#else
    // synchronous fallback: VGPR-staged copies
    for (int k0 = 0; k0 < K; k0 += GEMM_BK) {
        const unsigned short* gA = &A[(size_t)(m0 + lrow) * K + k0 + lseg];
        const unsigned short* gW = &W[(size_t)(n0 + lrow) * K + k0 + lseg];
        unsigned short* lA = &sA[0][lrow * LDS_STR + lseg];
        unsigned short* lW = &sW[0][lrow * LDS_STR + lseg];
        #pragma unroll
        for (int s = 0; s < 4; ++s) {
            *(uint4*)&lA[s * 8] = *(const uint4*)&gA[s * 8];
            *(uint4*)&lW[s * 8] = *(const uint4*)&gW[s * 8];
        }
        __syncthreads();
        compute_tile(0);
        __syncthreads();
    }
#endif

    // C/D layout: lane n = lane&15; VGPR r -> M = r + 8*(lane>>4)
    const int cn   = lane & 15;
    const int rofs = (lane >> 4) * 8;
    #pragma unroll
    for (int ms = 0; ms < 2; ++ms) {
        #pragma unroll
        for (int ns = 0; ns < 4; ++ns) {
            int n = n0 + nw * 64 + ns * 16 + cn;
            float bv = bias ? bias[n] : 0.0f;
            #pragma unroll
            for (int r = 0; r < 8; ++r) {
                int m = m0 + mw * 32 + ms * 16 + r + rofs;
                C[(size_t)m * N + n] = acc[ms][ns][r] + bv;
            }
        }
    }
}

// ---------------------------------------------------------------------------
// Causal depthwise conv1d + bias + SiLU; writes f32 and bf16 copies.
// grid: (D_INNER/256, M_ROWS)
// ---------------------------------------------------------------------------
__global__ void conv_silu_kernel(const float* __restrict__ xz,     // [M, 8192]
                                 const float* __restrict__ conv_w, // [D_INNER,1,4]
                                 const float* __restrict__ conv_b, // [D_INNER]
                                 float* __restrict__ xconv,        // [M, 4096]
                                 unsigned short* __restrict__ xconv_bf) {
    int d = blockIdx.x * 256 + threadIdx.x;
    int m = blockIdx.y;
    int l = m & (SEQ - 1);
    int bbase = (m - l) * (2 * D_INNER);    // (b*SEQ)*8192
    float s = conv_b[d];
    #pragma unroll
    for (int j = 0; j < D_CONV; ++j) {
        int ls = l - (D_CONV - 1) + j;
        if (ls >= 0)
            s += conv_w[d * D_CONV + j] * xz[(size_t)bbase + (size_t)ls * (2 * D_INNER) + d];
    }
    float v = silu_f(s);
    size_t o = (size_t)m * D_INNER + d;
    xconv[o] = v;
    xconv_bf[o] = f2bf(v);
}

// ---------------------------------------------------------------------------
// softplus in place
// ---------------------------------------------------------------------------
__global__ void softplus_kernel(float* __restrict__ x, int n) {
    int i = blockIdx.x * 256 + threadIdx.x;
    if (i < n) {
        float v = x[i];
        x[i] = (v > 20.0f) ? v : log1pf(__expf(v));
    }
}

// ---------------------------------------------------------------------------
// Selective scan: one thread per (b, d) channel; 16 states in registers.
// ---------------------------------------------------------------------------
__global__ __launch_bounds__(256) void scan_kernel(
    const float* __restrict__ dt,      // [M, D_INNER] (post-softplus)
    const float* __restrict__ BC,      // [M, 128]: cols 0..15 = B, 16..31 = C
    const float* __restrict__ xconv,   // [M, D_INNER]
    const float* __restrict__ A_log,   // [D_INNER, 16]
    const float* __restrict__ D_param, // [D_INNER]
    float* __restrict__ ys)            // [M, D_INNER]
{
    int idx = blockIdx.x * 256 + threadIdx.x;   // 0 .. B_SZ*D_INNER-1
    int b = idx >> 12;
    int d = idx & (D_INNER - 1);

    float A[D_STATE];
    #pragma unroll
    for (int n = 0; n < D_STATE; ++n) A[n] = -__expf(A_log[d * D_STATE + n]);
    float Dp = D_param[d];

    float h[D_STATE];
    #pragma unroll
    for (int n = 0; n < D_STATE; ++n) h[n] = 0.0f;

    for (int l = 0; l < SEQ; ++l) {
        int m = (b << 10) + l;
        size_t o = (size_t)m * D_INNER + d;
        float dtv = dt[o];
        float xv  = xconv[o];
        const float* bc = &BC[(size_t)m * BC_STRIDE];
        float y   = Dp * xv;
        float xdt = xv * dtv;
        #pragma unroll
        for (int n = 0; n < D_STATE; ++n) {
            float dA = __expf(dtv * A[n]);
            h[n] = h[n] * dA + xdt * bc[n];
            y += h[n] * bc[16 + n];
        }
        ys[o] = y;
    }
}

// ---------------------------------------------------------------------------
// y = ys * silu(z)  -> bf16 for the output GEMM
// ---------------------------------------------------------------------------
__global__ void gate_kernel(const float* __restrict__ ys,   // [M, D_INNER]
                            const float* __restrict__ xz,   // [M, 8192]
                            unsigned short* __restrict__ ybf) {
    int i = blockIdx.x * 256 + threadIdx.x;     // M * D_INNER
    int m = i >> 12;
    int d = i & (D_INNER - 1);
    float z = xz[(size_t)m * (2 * D_INNER) + D_INNER + d];
    ybf[i] = f2bf(ys[i] * silu_f(z));
}

// ---------------------------------------------------------------------------
// launch
// ---------------------------------------------------------------------------
extern "C" void kernel_launch(void* const* d_in, const int* in_sizes, int n_in,
                              void* d_out, int out_size, void* d_ws, size_t ws_size,
                              hipStream_t stream) {
    const float* u       = (const float*)d_in[0];
    const float* w_in    = (const float*)d_in[1];
    const float* b_in    = (const float*)d_in[2];
    const float* w_out   = (const float*)d_in[3];
    const float* b_out   = (const float*)d_in[4];
    const float* w_dt    = (const float*)d_in[5];
    const float* b_dt    = (const float*)d_in[6];
    const float* w_xdt   = (const float*)d_in[7];
    const float* w_xb    = (const float*)d_in[8];
    const float* w_xc    = (const float*)d_in[9];
    const float* conv_w  = (const float*)d_in[10];
    const float* conv_b  = (const float*)d_in[11];
    const float* A_log   = (const float*)d_in[12];
    const float* D_param = (const float*)d_in[13];
    float* out = (float*)d_out;

    // workspace carve-up (256B aligned)
    char* p = (char*)d_ws;
    size_t off = 0;
    auto alloc = [&](size_t bytes) -> void* {
        void* r = p + off;
        off += (bytes + 255) & ~(size_t)255;
        return r;
    };
    unsigned short* u_bf     = (unsigned short*)alloc((size_t)M_ROWS * D_MODEL * 2);
    unsigned short* w_in_bf  = (unsigned short*)alloc((size_t)2 * D_INNER * D_MODEL * 2);
    unsigned short* w_out_bf = (unsigned short*)alloc((size_t)D_MODEL * D_INNER * 2);
    unsigned short* w_dt_bf  = (unsigned short*)alloc((size_t)D_INNER * DT_RANK * 2);
    unsigned short* w_xdt_bf = (unsigned short*)alloc((size_t)DT_RANK * D_INNER * 2);
    unsigned short* wbc_bf   = (unsigned short*)alloc((size_t)BC_STRIDE * D_INNER * 2);
    float*          xz       = (float*)alloc((size_t)M_ROWS * 2 * D_INNER * 4);
    float*          xconv    = (float*)alloc((size_t)M_ROWS * D_INNER * 4);
    unsigned short* xconv_bf = (unsigned short*)alloc((size_t)M_ROWS * D_INNER * 2);
    float*          BC       = (float*)alloc((size_t)M_ROWS * BC_STRIDE * 4);
    float*          dtr      = (float*)alloc((size_t)M_ROWS * DT_RANK * 4);
    unsigned short* dtr_bf   = (unsigned short*)alloc((size_t)M_ROWS * DT_RANK * 2);
    float*          dt       = (float*)alloc((size_t)M_ROWS * D_INNER * 4);
    float*          ys       = (float*)alloc((size_t)M_ROWS * D_INNER * 4);
    unsigned short* ybf      = (unsigned short*)alloc((size_t)M_ROWS * D_INNER * 2);

    const dim3 blk(256);
    auto gsz = [](size_t n) { return dim3((unsigned)((n + 255) / 256)); };

    // 1. bf16 conversions
    cvt_bf16_kernel<<<gsz((size_t)M_ROWS * D_MODEL), blk, 0, stream>>>(u, u_bf, M_ROWS * D_MODEL);
    cvt_bf16_kernel<<<gsz((size_t)2 * D_INNER * D_MODEL), blk, 0, stream>>>(w_in, w_in_bf, 2 * D_INNER * D_MODEL);
    cvt_bf16_kernel<<<gsz((size_t)D_MODEL * D_INNER), blk, 0, stream>>>(w_out, w_out_bf, D_MODEL * D_INNER);
    cvt_bf16_kernel<<<gsz((size_t)D_INNER * DT_RANK), blk, 0, stream>>>(w_dt, w_dt_bf, D_INNER * DT_RANK);
    cvt_bf16_kernel<<<gsz((size_t)DT_RANK * D_INNER), blk, 0, stream>>>(w_xdt, w_xdt_bf, DT_RANK * D_INNER);
    build_wbc_kernel<<<gsz((size_t)BC_STRIDE * D_INNER), blk, 0, stream>>>(w_xb, w_xc, wbc_bf);

    // 2. xz = u @ w_in^T + b_in   [2048 x 8192], K=2048
    wmma_gemm_bf16<<<dim3((2 * D_INNER) / 128, M_ROWS / 128), blk, 0, stream>>>(
        u_bf, w_in_bf, b_in, xz, M_ROWS, 2 * D_INNER, D_MODEL);

    // 3. causal depthwise conv + SiLU
    conv_silu_kernel<<<dim3(D_INNER / 256, M_ROWS), blk, 0, stream>>>(
        xz, conv_w, conv_b, xconv, xconv_bf);

    // 4. BC = xconv @ [w_xb;w_xc;0]^T   [2048 x 128], K=4096
    wmma_gemm_bf16<<<dim3(1, M_ROWS / 128), blk, 0, stream>>>(
        xconv_bf, wbc_bf, nullptr, BC, M_ROWS, BC_STRIDE, D_INNER);

    // 5. dtr = xconv @ w_xdt^T   [2048 x 128], K=4096
    wmma_gemm_bf16<<<dim3(DT_RANK / 128, M_ROWS / 128), blk, 0, stream>>>(
        xconv_bf, w_xdt_bf, nullptr, dtr, M_ROWS, DT_RANK, D_INNER);
    cvt_bf16_kernel<<<gsz((size_t)M_ROWS * DT_RANK), blk, 0, stream>>>(dtr, dtr_bf, M_ROWS * DT_RANK);

    // 6. dt = softplus(dtr @ w_dt^T + b_dt)   [2048 x 4096], K=128
    wmma_gemm_bf16<<<dim3(D_INNER / 128, M_ROWS / 128), blk, 0, stream>>>(
        dtr_bf, w_dt_bf, b_dt, dt, M_ROWS, D_INNER, DT_RANK);
    softplus_kernel<<<gsz((size_t)M_ROWS * D_INNER), blk, 0, stream>>>(dt, M_ROWS * D_INNER);

    // 7. selective scan
    scan_kernel<<<dim3((B_SZ * D_INNER) / 256), blk, 0, stream>>>(
        dt, BC, xconv, A_log, D_param, ys);

    // 8. gate: y = ys * silu(z) -> bf16
    gate_kernel<<<gsz((size_t)M_ROWS * D_INNER), blk, 0, stream>>>(ys, xz, ybf);

    // 9. out = y @ w_out^T + b_out   [2048 x 2048], K=4096
    wmma_gemm_bf16<<<dim3(D_MODEL / 128, M_ROWS / 128), blk, 0, stream>>>(
        ybf, w_out_bf, b_out, out, M_ROWS, D_MODEL, D_INNER);
}
